// AttentionHead_45973329936973
// MI455X (gfx1250) — matrix operand order repaired and checked by reading, not compiled, not run on hardware
//
#include <hip/hip_runtime.h>

#define BB 4
#define TT 4096
#define DD 256
#define HH 64
#define KT 64            // attention key-tile size

typedef __attribute__((ext_vector_type(16))) _Float16     v16h;
typedef __attribute__((ext_vector_type(8)))  _Float16     v8h;
typedef __attribute__((ext_vector_type(8)))  float        v8f;
typedef __attribute__((ext_vector_type(4)))  unsigned int u32x4;
typedef __attribute__((ext_vector_type(8)))  int          i32x8;
typedef __attribute__((ext_vector_type(4)))  int          i32x4;

static __device__ __forceinline__ v8f wmma_f16(v16h a, v16h b, v8f c) {
    // D = A(16x32 f16) * B(32x16 f16) + C(16x16 f32)
    return __builtin_amdgcn_wmma_f32_16x16x32_f16(false, a, false, b, (short)0, c,
                                                  false, false);
}

// ---------------------------------------------------------------------------
// TDM: 2-D tile load global -> LDS. Builds D# groups per CDNA5 ISA 08 §8.
// data_size = 2 bytes. tile_d0 = contiguous extent, tile_d1 = #lines,
// stride0 = line stride (elements).
// ---------------------------------------------------------------------------
static __device__ __forceinline__ void tdm_load_2d(unsigned lds_off,
                                                   unsigned long long gaddr,
                                                   unsigned tile_d0, unsigned tile_d1,
                                                   unsigned long long stride0) {
    u32x4 g0;
    g0[0] = 1u;                                        // count=1, user descriptor
    g0[1] = lds_off;                                   // lds_addr   [63:32]
    g0[2] = (unsigned)(gaddr & 0xffffffffull);         // global_addr[95:64]
    g0[3] = (unsigned)((gaddr >> 32) & 0x1ffffffull)   // global_addr[120:96]
          | (2u << 30);                                // type=2 (image)
    i32x8 g1;
    g1[0] = (int)(1u << 16);                           // data_size=2B; mask=0
    g1[1] = (int)((tile_d0 & 0xffffu) << 16);          // tensor_dim0[15:0] @ bits63:48
    g1[2] = (int)((tile_d0 >> 16) | ((tile_d1 & 0xffffu) << 16)); // dim0 hi | dim1 lo
    g1[3] = (int)((tile_d1 >> 16) | (tile_d0 << 16));  // dim1 hi | tile_dim0
    g1[4] = (int)(tile_d1 & 0xffffu);                  // tile_dim1; tile_dim2=0
    g1[5] = (int)(stride0 & 0xffffffffull);            // tensor_dim0_stride lo32
    g1[6] = (int)((stride0 >> 32) & 0xffffull);        // stride hi16; dim1_stride lo=0
    g1[7] = 0;
    i32x4 gz = {0, 0, 0, 0};
#if defined(__clang_major__) && __clang_major__ >= 23
    i32x8 gz8 = {0, 0, 0, 0, 0, 0, 0, 0};
    __builtin_amdgcn_tensor_load_to_lds(g0, g1, gz, gz, gz8, 0);
#else
    __builtin_amdgcn_tensor_load_to_lds(g0, g1, gz, gz, 0);
#endif
}

// ---------------------------------------------------------------------------
// Kernel 0: WT[p][n][k] f16 = W_p[k][n]  (B-fragment friendly layout)
// ---------------------------------------------------------------------------
__global__ __launch_bounds__(256)
void prep_w_kernel(const float* __restrict__ Wq, const float* __restrict__ Wk,
                   const float* __restrict__ Wv, _Float16* __restrict__ WT) {
    const float* W = (blockIdx.y == 0) ? Wq : (blockIdx.y == 1) ? Wk : Wv;
    const int g = blockIdx.x * 256 + threadIdx.x;     // 0..16383
    const int n = g >> 8;                             // 0..63
    const int k = g & 255;                            // 0..255
    WT[((size_t)blockIdx.y * HH + n) * DD + k] = (_Float16)W[(size_t)k * HH + n];
}

// ---------------------------------------------------------------------------
// Kernel 1: q = rope2d(x@Wq)/16 -> [b][t][h], k = rope2d(x@Wk) -> [b][t][h],
//           v = x@Wv stored TRANSPOSED -> vT[b][h][t].  All f16 in workspace.
// One wave per 16-row tile of the flattened B*T rows.
// Workspace layout: qf | kf | vT | WT  (N = B*T*H elements each for the first 3)
// ---------------------------------------------------------------------------
__global__ __launch_bounds__(256)
void proj_rope_kernel(const float* __restrict__ x,
                      const float* __restrict__ angx,
                      const float* __restrict__ angy,
                      _Float16* __restrict__ ws16) {
    const size_t N = (size_t)BB * TT * HH;
    _Float16* vT       = ws16 + 2 * N;
    const _Float16* WT = ws16 + 3 * N;

    __shared__ __attribute__((aligned(32))) float ltile[8 * 256];

    const int wave = threadIdx.x >> 5;
    const int lane = threadIdx.x & 31;
    const int half = lane >> 4;
    const int ln   = lane & 15;

    const int tile = blockIdx.x * 8 + wave;       // 16-row tile index (0..1023)
    const float* xr = x + (size_t)(tile * 16 + ln) * DD;

    // Preload 8 A-fragments (x tile 16x256 -> 8 x (16x32 f16)).
    // A layout: lane<16: K = {0..15}; lane>=16: K = {16..31}
    v16h a[8];
    #pragma unroll
    for (int ks = 0; ks < 8; ++ks) {
        const int b0 = ks * 32 + 16 * half;
        #pragma unroll
        for (int j = 0; j < 8; ++j) a[ks][j]     = (_Float16)xr[b0 + j];
        #pragma unroll
        for (int j = 0; j < 8; ++j) a[ks][8 + j] = (_Float16)xr[b0 + 8 + j];
    }

    float* mytile = ltile + wave * 256;

    for (int p = 0; p < 3; ++p) {
        #pragma unroll
        for (int nt = 0; nt < 4; ++nt) {
            const _Float16* wrow = WT + (size_t)(p * HH + nt * 16 + ln) * DD + 16 * half;
            // Preload all 8 B-fragments, then run the WMMA chain back-to-back.
            v16h bfr[8];
            #pragma unroll
            for (int ks = 0; ks < 8; ++ks)
                bfr[ks] = *(const v16h*)(wrow + ks * 32);
            v8f acc = {};
            #pragma unroll
            for (int ks = 0; ks < 8; ++ks)
                acc = wmma_f16(a[ks], bfr[ks], acc);

            // C layout -> LDS so lanes can re-slice row-contiguously.
            #pragma unroll
            for (int r = 0; r < 8; ++r)
                mytile[(r + 8 * half) * 16 + ln] = acc[r];
            __syncthreads();

            if (p < 2) {
                // lane: row = ln, cols [8*half, 8*half+8) of this 16-col tile
                float vals[8];
                #pragma unroll
                for (int j = 0; j < 8; ++j) vals[j] = mytile[ln * 16 + 8 * half + j];

                const int trow = tile * 16 + ln;           // flattened B*T row
                const int t    = trow & (TT - 1);
                const int c0   = nt * 16 + 8 * half;       // head-dim col of vals[0]
                #pragma unroll
                for (int pr = 0; pr < 4; ++pr) {           // 2D RoPE
                    const int c = c0 + 2 * pr;
                    const float ang = (c < 32) ? angx[t * 16 + (c >> 1)]
                                               : angy[t * 16 + ((c - 32) >> 1)];
                    float sn, cs;
                    __sincosf(ang, &sn, &cs);
                    const float va = vals[2 * pr], vb = vals[2 * pr + 1];
                    vals[2 * pr]     = va * cs - vb * sn;
                    vals[2 * pr + 1] = va * sn + vb * cs;
                }
                const float mul = (p == 0) ? 0.0625f : 1.0f;  // fold 1/sqrt(256)
                v8h h8;
                #pragma unroll
                for (int j = 0; j < 8; ++j) h8[j] = (_Float16)(vals[j] * mul);
                *(v8h*)(ws16 + (size_t)p * N + (size_t)trow * HH + c0) = h8;
            } else {
                // v: store transposed. lane: h-col = ln of tile, t-chunk = 8*half
                const int h    = nt * 16 + ln;
                const int bidx = tile >> 8;                // 256 tiles per batch
                const int t0   = ((tile & 255) * 16) + 8 * half;
                v8h h8;
                #pragma unroll
                for (int j = 0; j < 8; ++j)
                    h8[j] = (_Float16)mytile[(8 * half + j) * 16 + ln];
                *(v8h*)(vT + ((size_t)bidx * HH + h) * TT + t0) = h8;
            }
            __syncthreads();
        }
    }
}

// ---------------------------------------------------------------------------
// Kernel 2: flash attention. 4 waves/block, 16 query rows per wave, 64-key
// tiles. K and V^T tiles streamed into double-buffered LDS by the TDM so the
// tensor engine overlaps the WMMA pipeline.
// ---------------------------------------------------------------------------
__global__ __launch_bounds__(128)
void flash_attn_kernel(const _Float16* __restrict__ ws16, float* __restrict__ out) {
    const size_t N = (size_t)BB * TT * HH;
    const _Float16* qf = ws16;
    const _Float16* kf = ws16 + N;
    const _Float16* vT = ws16 + 2 * N;

    __shared__ __attribute__((aligned(64))) _Float16 kt[2][KT * HH];   // [key][h]
    __shared__ __attribute__((aligned(64))) _Float16 vtT[2][HH * KT];  // [h][key]
    __shared__ __attribute__((aligned(32))) _Float16 pt[4 * 16 * KT];

    const int tid  = threadIdx.x;
    const int wave = tid >> 5;
    const int lane = tid & 31;
    const int half = lane >> 4;
    const int ln   = lane & 15;

    const int b    = blockIdx.x >> 6;            // T/64 = 64 q-blocks per batch
    const int qblk = (blockIdx.x & 63) * 64;

    // Q A-fragments: 16 rows x 64 -> two K=32 fragments.
    const _Float16* qp = qf + ((size_t)b * TT + qblk + wave * 16 + ln) * HH;
    v16h qa[2];
    #pragma unroll
    for (int s = 0; s < 2; ++s) {
        v8h lo = *(const v8h*)(qp + 32 * s + 16 * half);
        v8h hi = *(const v8h*)(qp + 32 * s + 16 * half + 8);
        qa[s] = __builtin_shufflevector(lo, hi, 0, 1, 2, 3, 4, 5, 6, 7,
                                        8, 9, 10, 11, 12, 13, 14, 15);
    }

    v8f o[4] = {};
    float m[8], l[8];
    #pragma unroll
    for (int r = 0; r < 8; ++r) { m[r] = -__builtin_inff(); l[r] = 0.0f; }

    _Float16* pw = pt + wave * (16 * KT);
    const unsigned kt_off = (unsigned)(size_t)(void*)&kt[0][0];
    const unsigned vt_off = (unsigned)(size_t)(void*)&vtT[0][0];
    const unsigned long long kgbase = (unsigned long long)(size_t)(kf + (size_t)b * TT * HH);
    const unsigned long long vgbase = (unsigned long long)(size_t)(vT + (size_t)b * HH * TT);

    constexpr int NIT = TT / KT;                 // 64 key tiles

    // Preload tile 0 into buffer 0.
    if (wave == 0) {
        tdm_load_2d(kt_off, kgbase, (unsigned)HH, (unsigned)KT, (unsigned long long)HH);
        tdm_load_2d(vt_off, vgbase, (unsigned)KT, (unsigned)HH, (unsigned long long)TT);
    }

    for (int it = 0; it < NIT; ++it) {
        if (wave == 0) {
            if (it + 1 < NIT) {
                const int nb = (it + 1) & 1;
                const unsigned long long koff = (unsigned long long)(it + 1) * KT;
                tdm_load_2d(kt_off + nb * (KT * HH * 2),
                            kgbase + koff * HH * 2,
                            (unsigned)HH, (unsigned)KT, (unsigned long long)HH);
                tdm_load_2d(vt_off + nb * (HH * KT * 2),
                            vgbase + koff * 2,
                            (unsigned)KT, (unsigned)HH, (unsigned long long)TT);
                __builtin_amdgcn_s_wait_tensorcnt(2);   // tile `it` has landed
            } else {
                __builtin_amdgcn_s_wait_tensorcnt(0);
            }
        }
        __syncthreads();

        const _Float16* ktb = &kt[it & 1][0];
        const _Float16* vtb = &vtT[it & 1][0];

        // S = Q K^T: 16 queries x 64 keys, contraction over H=64 in 2 steps.
        v8f s4[4] = {};
        #pragma unroll
        for (int s = 0; s < 2; ++s) {
            #pragma unroll
            for (int nt = 0; nt < 4; ++nt) {
                v16h kbf = *(const v16h*)(ktb + (nt * 16 + ln) * HH + 32 * s + 16 * half);
                s4[nt] = wmma_f16(qa[s], kbf, s4[nt]);
            }
        }

        // Online softmax (rows live on 16-lane N-groups of the C layout).
        #pragma unroll
        for (int r = 0; r < 8; ++r) {
            float mx = fmaxf(fmaxf(s4[0][r], s4[1][r]), fmaxf(s4[2][r], s4[3][r]));
            #pragma unroll
            for (int msk = 1; msk <= 8; msk <<= 1)
                mx = fmaxf(mx, __shfl_xor(mx, msk, 32));
            const float mn    = fmaxf(m[r], mx);
            const float alpha = __expf(m[r] - mn);
            m[r] = mn;
            float rs = 0.0f;
            #pragma unroll
            for (int nt = 0; nt < 4; ++nt) {
                const float p = __expf(s4[nt][r] - mn);
                s4[nt][r] = p;
                rs += p;
            }
            #pragma unroll
            for (int msk = 1; msk <= 8; msk <<= 1)
                rs += __shfl_xor(rs, msk, 32);
            l[r] = l[r] * alpha + rs;
            #pragma unroll
            for (int nt = 0; nt < 4; ++nt) o[nt][r] *= alpha;
        }

        // P (C layout, f32) -> per-wave LDS as f16 (16 x 64).
        #pragma unroll
        for (int r = 0; r < 8; ++r) {
            const int row = r + 8 * half;
            #pragma unroll
            for (int nt = 0; nt < 4; ++nt)
                pw[row * KT + nt * 16 + ln] = (_Float16)s4[nt][r];
        }
        __syncthreads();

        // P as two K=32 A-fragments; V from transposed LDS tile.
        #pragma unroll
        for (int s = 0; s < 2; ++s) {
            v8h plo = *(const v8h*)(pw + ln * KT + 32 * s + 16 * half);
            v8h phi = *(const v8h*)(pw + ln * KT + 32 * s + 16 * half + 8);
            v16h pa = __builtin_shufflevector(plo, phi, 0, 1, 2, 3, 4, 5, 6, 7,
                                              8, 9, 10, 11, 12, 13, 14, 15);
            #pragma unroll
            for (int ht = 0; ht < 4; ++ht) {
                v16h vb = *(const v16h*)(vtb + (ht * 16 + ln) * KT + 32 * s + 16 * half);
                o[ht] = wmma_f16(pa, vb, o[ht]);
            }
        }
        __syncthreads();   // all waves done with this buffer pair
    }

    // Normalize and store (f32, C-layout mapping).
    #pragma unroll
    for (int ht = 0; ht < 4; ++ht) {
        #pragma unroll
        for (int r = 0; r < 8; ++r) {
            const int t = qblk + wave * 16 + r + 8 * half;
            out[((size_t)b * TT + t) * HH + ht * 16 + ln] = o[ht][r] / l[r];
        }
    }
}

// ---------------------------------------------------------------------------
extern "C" void kernel_launch(void* const* d_in, const int* in_sizes, int n_in,
                              void* d_out, int out_size, void* d_ws, size_t ws_size,
                              hipStream_t stream) {
    const float* x  = (const float*)d_in[0];
    const float* Wq = (const float*)d_in[1];
    const float* Wk = (const float*)d_in[2];
    const float* Wv = (const float*)d_in[3];
    const float* ax = (const float*)d_in[4];
    const float* ay = (const float*)d_in[5];

    const size_t N = (size_t)BB * TT * HH;
    _Float16* ws16 = (_Float16*)d_ws;   // qf | kf | vT | WT
    _Float16* WT   = ws16 + 3 * N;

    prep_w_kernel<<<dim3(64, 3), dim3(256), 0, stream>>>(Wq, Wk, Wv, WT);
    proj_rope_kernel<<<dim3(BB * TT / 128), dim3(256), 0, stream>>>(x, ax, ay, ws16);
    flash_attn_kernel<<<dim3(BB * (TT / 64)), dim3(128), 0, stream>>>(ws16, (float*)d_out);
}
